// Float8BMM_34720515621131
// MI455X (gfx1250) — 8-bit matrix operand order repaired and checked by reading, compiled, not run on hardware
//
#include <hip/hip_runtime.h>

typedef __attribute__((ext_vector_type(16))) int   v16i;
typedef __attribute__((ext_vector_type(4)))  int   v4i;
typedef __attribute__((ext_vector_type(2)))  int   v2i;
typedef __attribute__((ext_vector_type(8)))  float v8f;

#define BATCH 8
#define MDIM  2048
#define NDIM  2048
#define KDIM  2048
#define FP8MAX 448.0f
#define EPSF   1.1920928955078125e-07f

__device__ inline unsigned char f32_to_e4m3_sw(float f) {
    f = fminf(fmaxf(f, -FP8MAX), FP8MAX);
    unsigned u = __float_as_uint(f);
    unsigned s = (u >> 24) & 0x80u;
    u &= 0x7fffffffu;
    if (u < 0x3C800000u) {
        float a = __uint_as_float(u);
        unsigned d = (unsigned)(a * 512.0f + 0.5f);
        return (unsigned char)(s | d);
    }
    u += 0x00080000u;
    unsigned exp  = (u >> 23) & 0xFFu;
    unsigned mant = (u >> 20) & 0x7u;
    int e = (int)exp - 127 + 7;
    if (e > 15 || (e == 15 && mant == 7)) return (unsigned char)(s | 0x7E);
    if (e <= 0) return (unsigned char)s;
    return (unsigned char)(s | ((unsigned)e << 3) | mant);
}

__device__ inline unsigned pack4_fp8(float a, float b, float c, float d) {
    a = fminf(fmaxf(a, -FP8MAX), FP8MAX);
    b = fminf(fmaxf(b, -FP8MAX), FP8MAX);
    c = fminf(fmaxf(c, -FP8MAX), FP8MAX);
    d = fminf(fmaxf(d, -FP8MAX), FP8MAX);
#if __has_builtin(__builtin_amdgcn_cvt_pk_fp8_f32)
    int v = 0;
    v = __builtin_amdgcn_cvt_pk_fp8_f32(a, b, v, false);
    v = __builtin_amdgcn_cvt_pk_fp8_f32(c, d, v, true);
    return (unsigned)v;
#else
    return (unsigned)f32_to_e4m3_sw(a)
         | ((unsigned)f32_to_e4m3_sw(b) << 8)
         | ((unsigned)f32_to_e4m3_sw(c) << 16)
         | ((unsigned)f32_to_e4m3_sw(d) << 24);
#endif
}

__global__ void init_hdr_kernel(unsigned* hdr_u) {
    hdr_u[0] = 0u;
    hdr_u[1] = 0u;
}

__global__ __launch_bounds__(256)
void amax_kernel(const float4* __restrict__ src, unsigned n4, unsigned* __restrict__ dst) {
    __shared__ float red[256];
    float m = 0.0f;
    for (unsigned i = blockIdx.x * blockDim.x + threadIdx.x; i < n4;
         i += gridDim.x * blockDim.x) {
        float4 v = src[i];
        m = fmaxf(m, fabsf(v.x));
        m = fmaxf(m, fabsf(v.y));
        m = fmaxf(m, fabsf(v.z));
        m = fmaxf(m, fabsf(v.w));
    }
    red[threadIdx.x] = m;
    __syncthreads();
    for (int s = 128; s > 0; s >>= 1) {
        if ((int)threadIdx.x < s)
            red[threadIdx.x] = fmaxf(red[threadIdx.x], red[threadIdx.x + s]);
        __syncthreads();
    }
    if (threadIdx.x == 0)
        atomicMax(dst, __float_as_uint(red[0]));
}

__global__ void scale_kernel(const unsigned* __restrict__ hdr_u, float* __restrict__ hdr_f) {
    float ax = fmaxf(__uint_as_float(hdr_u[0]), EPSF);
    float aw = fmaxf(__uint_as_float(hdr_u[1]), EPSF);
    float sx = FP8MAX / ax;
    float sw = FP8MAX / aw;
    hdr_f[2] = sx;
    hdr_f[3] = sw;
    hdr_f[4] = 1.0f / (sx * sw);
}

__global__ __launch_bounds__(256)
void quant_x_kernel(const float4* __restrict__ x, const float* __restrict__ hdr_f,
                    unsigned* __restrict__ xq) {
    float s = hdr_f[2];
    size_t i = (size_t)blockIdx.x * blockDim.x + threadIdx.x;
    float4 v = x[i];
    xq[i] = pack4_fp8(v.x * s, v.y * s, v.z * s, v.w * s);
}

__global__ __launch_bounds__(256)
void quant_wT_kernel(const float* __restrict__ w, const float* __restrict__ hdr_f,
                     unsigned char* __restrict__ wqT) {
    __shared__ float tile[32][33];
    const int n0 = blockIdx.x * 32;
    const int k0 = blockIdx.y * 32;
    const int b  = blockIdx.z;

    const float* wp = w + ((size_t)b * KDIM + k0) * NDIM + n0;
#pragma unroll
    for (int i = 0; i < 4; ++i) {
        int kl = threadIdx.y + i * 8;
        tile[kl][threadIdx.x] = wp[(size_t)kl * NDIM + threadIdx.x];
    }
    __syncthreads();

    float s = hdr_f[3];
    int tid = threadIdx.y * 32 + threadIdx.x;
    int nl  = tid >> 3;
    int jc  = tid & 7;
    unsigned pk = pack4_fp8(tile[jc * 4 + 0][nl] * s,
                            tile[jc * 4 + 1][nl] * s,
                            tile[jc * 4 + 2][nl] * s,
                            tile[jc * 4 + 3][nl] * s);
    *(unsigned*)(wqT + ((size_t)b * NDIM + (n0 + nl)) * KDIM + k0 + jc * 4) = pk;
}

__global__ __launch_bounds__(256)
void gemm_fp8_kernel(const unsigned char* __restrict__ xq,
                     const unsigned char* __restrict__ wqT,
                     const float* __restrict__ bias,
                     float* __restrict__ out,
                     const float* __restrict__ hdr_f) {
    const int b     = blockIdx.z;
    const int nBase = blockIdx.x * 128;
    const int mBase = blockIdx.y * 128;
    const int wave  = threadIdx.x >> 5;
    const int lane  = threadIdx.x & 31;
    const int half  = lane >> 4;
    const int lr    = lane & 15;
    const int wm    = (wave >> 1) * 32;
    const int wn    = (wave & 1) * 64;

    const float inv = hdr_f[4];

    v8f acc[2][4];
#pragma unroll
    for (int mt = 0; mt < 2; ++mt)
#pragma unroll
        for (int nt = 0; nt < 4; ++nt)
            acc[mt][nt] = v8f{};

    const unsigned char* aRow[2];
    aRow[0] = xq + ((size_t)b * MDIM + (mBase + wm + lr)) * KDIM + half * 16;
    aRow[1] = aRow[0] + (size_t)16 * KDIM;

    const unsigned char* bCol[4];
#pragma unroll
    for (int nt = 0; nt < 4; ++nt)
        bCol[nt] = wqT + ((size_t)b * NDIM + (nBase + wn + nt * 16 + lr)) * KDIM + half * 16;

    for (int k0 = 0; k0 < KDIM; k0 += 128) {
        v16i A[2];
        v16i Bf[4];
#pragma unroll
        for (int t = 0; t < 2; ++t) {
            const unsigned char* p = aRow[t] + k0;
#pragma unroll
            for (int blk = 0; blk < 2; ++blk) {
#pragma unroll
                for (int q = 0; q < 4; ++q) {
                    v2i d = *(const v2i*)(p + blk * 64 + (q >> 1) * 32 + (q & 1) * 8);
                    A[t][blk * 8 + q * 2 + 0] = d[0];
                    A[t][blk * 8 + q * 2 + 1] = d[1];
                }
            }
        }
#pragma unroll
        for (int nt = 0; nt < 4; ++nt) {
            const unsigned char* p = bCol[nt] + k0;
#pragma unroll
            for (int q = 0; q < 4; ++q) {
                v4i d = *(const v4i*)(p + q * 32);
                Bf[nt][q * 4 + 0] = d[0];
                Bf[nt][q * 4 + 1] = d[1];
                Bf[nt][q * 4 + 2] = d[2];
                Bf[nt][q * 4 + 3] = d[3];
            }
        }
#pragma unroll
        for (int mt = 0; mt < 2; ++mt)
#pragma unroll
            for (int nt = 0; nt < 4; ++nt)
                acc[mt][nt] = __builtin_amdgcn_wmma_f32_16x16x128_fp8_fp8(
                    A[mt], Bf[nt], (short)0, acc[mt][nt], false, false);
    }

#pragma unroll
    for (int mt = 0; mt < 2; ++mt) {
#pragma unroll
        for (int nt = 0; nt < 4; ++nt) {
            int row0 = mBase + wm + mt * 16 + half * 8;
            int col  = nBase + wn + nt * 16 + lr;
            float bv = bias[(size_t)b * NDIM + col];
            float* op = out + ((size_t)b * MDIM + row0) * NDIM + col;
#pragma unroll
            for (int r = 0; r < 8; ++r)
                op[(size_t)r * NDIM] = acc[mt][nt][r] * inv + bv;
        }
    }
}

extern "C" void kernel_launch(void* const* d_in, const int* in_sizes, int n_in,
                              void* d_out, int out_size, void* d_ws, size_t ws_size,
                              hipStream_t stream) {
    (void)in_sizes; (void)n_in; (void)out_size; (void)ws_size;
    const float* x    = (const float*)d_in[0];
    const float* w    = (const float*)d_in[1];
    const float* bias = (const float*)d_in[2];
    float*       out  = (float*)d_out;

    unsigned char* ws    = (unsigned char*)d_ws;
    unsigned*      hdr_u = (unsigned*)ws;
    float*         hdr_f = (float*)ws;
    unsigned char* xq    = ws + 256;
    unsigned char* wqT   = xq + (size_t)BATCH * MDIM * KDIM;

    const size_t nElem = (size_t)BATCH * MDIM * KDIM;
    const unsigned n4  = (unsigned)(nElem / 4);

    init_hdr_kernel<<<1, 1, 0, stream>>>(hdr_u);
    amax_kernel<<<1024, 256, 0, stream>>>((const float4*)x, n4, hdr_u + 0);
    amax_kernel<<<1024, 256, 0, stream>>>((const float4*)w, n4, hdr_u + 1);
    scale_kernel<<<1, 1, 0, stream>>>(hdr_u, hdr_f);

    quant_x_kernel<<<n4 / 256, 256, 0, stream>>>((const float4*)x, hdr_f, (unsigned*)xq);

    dim3 qb(32, 8), qg(NDIM / 32, KDIM / 32, BATCH);
    quant_wT_kernel<<<qg, qb, 0, stream>>>(w, hdr_f, wqT);

    dim3 gg(NDIM / 128, MDIM / 128, BATCH), gb(256);
    gemm_fp8_kernel<<<gg, gb, 0, stream>>>(xq, wqT, bias, out, hdr_f);
}
